// LinAttentionalPropagation_89739046683497
// MI455X (gfx1250) — matrix operand order repaired and checked by reading, not compiled, not run on hardware
//
#include <hip/hip_runtime.h>
#include <hip/hip_bf16.h>

#define BB 4
#define DD 1024
#define NN 4096
#define HH 16
#define KK 128
#define DHD 64

typedef __attribute__((ext_vector_type(16))) __bf16 v16bf;
typedef __attribute__((ext_vector_type(8)))  float  v8f;

union Frag {
    v16bf v;
    uint4 q[2];
};

__device__ __forceinline__ unsigned f2bf(float f) {
    union { float f; unsigned u; } c;
    c.f = f;
    return c.u + 0x7FFFu + ((c.u >> 16) & 1u);   // result in high 16 bits (RNE)
}
__device__ __forceinline__ unsigned packbf(float lo, float hi) {
    return (f2bf(lo) >> 16) | (f2bf(hi) & 0xFFFF0000u);
}
__device__ __forceinline__ uint4 pack8(float4 a, float4 b) {
    uint4 u;
    u.x = packbf(a.x, a.y);
    u.y = packbf(a.z, a.w);
    u.z = packbf(b.x, b.y);
    u.w = packbf(b.z, b.w);
    return u;
}

// Generic batched GEMM: out[b, m, n] = sum_i A[b, m, i] * X[b, i, n] (+ bias[m]) (+ BN + ReLU)
// A: [M x Kc] row-major, batch stride a_bstride (0 => shared weights).
// X row i from X0 (i < split) or X1 (row i - split); both batch stride x_bstride.
// WG 256 threads (8 waves); WG tile 64(M) x 128(N); wave tile 16(M) x 64(N) = 4 accumulators.
// K staged in chunks of 64 through LDS (bf16), two 32-k WMMA sub-steps per stage.
template <int HAS_BIAS, int BNRELU, int SPLIT_EN>
__global__ __launch_bounds__(256)
void gemm_wmma_kernel(const float* __restrict__ A, size_t a_bstride,
                      const float* __restrict__ X0, const float* __restrict__ X1, int split,
                      size_t x_bstride,
                      const float* __restrict__ bias,
                      const float* __restrict__ bn_g, const float* __restrict__ bn_b,
                      const float* __restrict__ bn_m, const float* __restrict__ bn_v,
                      float* __restrict__ out, size_t o_bstride,
                      int Kc, int Ncols)
{
    // bf16 tiles; rows padded to 72 halves (144 B) => 16B-aligned fragments, bank spread
    __shared__ unsigned short lA[64 * 72];    // [m 0..63][k 0..63]
    __shared__ unsigned short lB[128 * 72];   // [n 0..127][k 0..63] (transposed X tile)

    const int t    = threadIdx.x;
    const int lane = t & 31;
    const int wave = t >> 5;
    const int wm   = wave & 3;        // 4 M-subtiles of 16
    const int wn   = wave >> 2;       // 2 N-groups of 64
    const int bidx = blockIdx.z;
    const int m0   = blockIdx.y * 64;
    const int n0   = blockIdx.x * 128;

    const float* Ab = A + (size_t)bidx * a_bstride;

    v8f c0 = {}, c1 = {}, c2 = {}, c3 = {};

    // A loader: thread -> (row, 16-k segment)
    const int arow = t >> 2, aseg = t & 3;
    // X loader: thread -> (even k-row pair, 16-n segment)
    const int kpair = t >> 3, xseg = t & 7;

    const int lnlo = lane & 15;
    const int kh   = lane >> 4;

    for (int kk = 0; kk < Kc; kk += 64) {
        // --- stage A tile: W[m0+arow, kk + aseg*16 .. +15] -> lA (2x ds_store_b128) ---
        {
            const float4* src = (const float4*)(Ab + (size_t)(m0 + arow) * (size_t)Kc + kk + aseg * 16);
            float4 f0 = src[0], f1 = src[1], f2 = src[2], f3 = src[3];
            uint4* d = (uint4*)&lA[arow * 72 + aseg * 16];
            d[0] = pack8(f0, f1);
            d[1] = pack8(f2, f3);
        }
        // --- stage X tile transposed: rows kk+2*kpair, kk+2*kpair+1; cols n0+xseg*16..+15 ---
        {
            const int gk0 = kk + 2 * kpair;
            const int gk1 = gk0 + 1;
            const float* r0;
            const float* r1;
            if (SPLIT_EN) {
                r0 = (gk0 < split) ? (X0 + (size_t)bidx * x_bstride + (size_t)gk0 * (size_t)Ncols)
                                   : (X1 + (size_t)bidx * x_bstride + (size_t)(gk0 - split) * (size_t)Ncols);
                r1 = (gk1 < split) ? (X0 + (size_t)bidx * x_bstride + (size_t)gk1 * (size_t)Ncols)
                                   : (X1 + (size_t)bidx * x_bstride + (size_t)(gk1 - split) * (size_t)Ncols);
            } else {
                r0 = X0 + (size_t)bidx * x_bstride + (size_t)gk0 * (size_t)Ncols;
                r1 = X0 + (size_t)bidx * x_bstride + (size_t)gk1 * (size_t)Ncols;
            }
            const float4* p0 = (const float4*)(r0 + n0 + xseg * 16);
            const float4* p1 = (const float4*)(r1 + n0 + xseg * 16);
            unsigned* db = (unsigned*)&lB[0];
            #pragma unroll
            for (int s = 0; s < 4; ++s) {
                float4 a = p0[s];
                float4 b = p1[s];
                const int nbase = xseg * 16 + s * 4;
                // lB half-index n*72 + 2*kpair, packed pair (k even, k odd) => one u32
                db[((nbase + 0) * 72 + 2 * kpair) >> 1] = packbf(a.x, b.x);
                db[((nbase + 1) * 72 + 2 * kpair) >> 1] = packbf(a.y, b.y);
                db[((nbase + 2) * 72 + 2 * kpair) >> 1] = packbf(a.z, b.z);
                db[((nbase + 3) * 72 + 2 * kpair) >> 1] = packbf(a.w, b.w);
            }
        }
        __syncthreads();

        // --- 2 sub-steps of K=32, 4 WMMAs each ---
        #pragma unroll
        for (int ks = 0; ks < 64; ks += 32) {
            Frag a;
            {
                // A 16x32: lane m = lnlo; elems 0..15 -> K=ks+kh*16+e
                const uint4* pa = (const uint4*)&lA[(wm * 16 + lnlo) * 72];
                const int b0 = ks >> 3;             // 0 or 4
                a.q[0] = pa[b0 + 2 * kh];
                a.q[1] = pa[b0 + 2 * kh + 1];
            }
            Frag bf;
            const int bb = (ks >> 3) + 2 * kh;      // uint4 index of 32B K-run
            {
                const uint4* pb = (const uint4*)&lB[(wn * 64 + 0 + lnlo) * 72];
                bf.q[0] = pb[bb]; bf.q[1] = pb[bb + 1];
                c0 = __builtin_amdgcn_wmma_f32_16x16x32_bf16(false, a.v, false, bf.v, (short)0, c0, false, false);
            }
            {
                const uint4* pb = (const uint4*)&lB[(wn * 64 + 16 + lnlo) * 72];
                bf.q[0] = pb[bb]; bf.q[1] = pb[bb + 1];
                c1 = __builtin_amdgcn_wmma_f32_16x16x32_bf16(false, a.v, false, bf.v, (short)0, c1, false, false);
            }
            {
                const uint4* pb = (const uint4*)&lB[(wn * 64 + 32 + lnlo) * 72];
                bf.q[0] = pb[bb]; bf.q[1] = pb[bb + 1];
                c2 = __builtin_amdgcn_wmma_f32_16x16x32_bf16(false, a.v, false, bf.v, (short)0, c2, false, false);
            }
            {
                const uint4* pb = (const uint4*)&lB[(wn * 64 + 48 + lnlo) * 72];
                bf.q[0] = pb[bb]; bf.q[1] = pb[bb + 1];
                c3 = __builtin_amdgcn_wmma_f32_16x16x32_bf16(false, a.v, false, bf.v, (short)0, c3, false, false);
            }
        }
        __syncthreads();
    }

    // --- epilogue: C layout: VGPR r, lane l: m = (l>>4)*8 + r, n = l&15 ---
    const int mrow = m0 + wm * 16 + kh * 8;
    const int ncol = n0 + wn * 64 + lnlo;
    float* ob = out + (size_t)bidx * o_bstride;
    #pragma unroll
    for (int r = 0; r < 8; ++r) {
        const int m = mrow + r;
        float v0 = c0[r], v1 = c1[r], v2 = c2[r], v3 = c3[r];
        if (HAS_BIAS) {
            const float add = bias[m];
            v0 += add; v1 += add; v2 += add; v3 += add;
        }
        if (BNRELU) {
            const float sc = bn_g[m] * rsqrtf(bn_v[m] + 1e-3f);
            const float sh = bn_b[m] - bn_m[m] * sc;
            v0 = fmaxf(v0 * sc + sh, 0.0f);
            v1 = fmaxf(v1 * sc + sh, 0.0f);
            v2 = fmaxf(v2 * sc + sh, 0.0f);
            v3 = fmaxf(v3 * sc + sh, 0.0f);
        }
        float* orow = ob + (size_t)m * (size_t)Ncols + ncol;
        orow[0]  = v0;
        orow[16] = v1;
        orow[32] = v2;
        orow[48] = v3;
    }
}

// Attention: per (b, h, n): scores[k] = (1/8) * sum_dh q[b, dh*16+h, n] * kp[b, dh*16+h, k],
// softmax over k (K=128), msg[b, dh*16+h, n] = sum_k prob[k] * kp[b, dh*16+h, k]   (v == k).
__global__ __launch_bounds__(256)
void attn_kernel(const float* __restrict__ q, const float* __restrict__ kp,
                 float* __restrict__ msg)
{
    __shared__ float kh[DHD * KK];   // 32 KB, broadcast reads
    const int b = blockIdx.z;
    const int h = blockIdx.y;
    const int n = blockIdx.x * 256 + threadIdx.x;

    for (int i = threadIdx.x; i < DHD * KK; i += 256) {
        const int dh = i >> 7, k = i & 127;
        kh[i] = kp[(size_t)b * DD * KK + (size_t)(dh * HH + h) * KK + k];
    }
    __syncthreads();

    float qv[DHD];
    #pragma unroll
    for (int dh = 0; dh < DHD; ++dh)
        qv[dh] = q[(size_t)b * DD * NN + (size_t)(dh * HH + h) * NN + n] * 0.125f; // 1/sqrt(64)

    // pass 1: row max
    float mx = -3.4e38f;
    for (int k = 0; k < KK; k += 4) {
        float s0 = 0.f, s1 = 0.f, s2 = 0.f, s3 = 0.f;
        #pragma unroll
        for (int dh = 0; dh < DHD; ++dh) {
            const float* kr = &kh[dh * KK + k];
            s0 += qv[dh] * kr[0]; s1 += qv[dh] * kr[1];
            s2 += qv[dh] * kr[2]; s3 += qv[dh] * kr[3];
        }
        mx = fmaxf(mx, fmaxf(fmaxf(s0, s1), fmaxf(s2, s3)));
    }

    // pass 2: exp-normalize and accumulate P*V
    float acc[DHD];
    #pragma unroll
    for (int dh = 0; dh < DHD; ++dh) acc[dh] = 0.f;
    float den = 0.f;
    for (int k = 0; k < KK; ++k) {
        float s = 0.f;
        #pragma unroll
        for (int dh = 0; dh < DHD; ++dh) s += qv[dh] * kh[dh * KK + k];
        const float e = __expf(s - mx);
        den += e;
        #pragma unroll
        for (int dh = 0; dh < DHD; ++dh) acc[dh] += e * kh[dh * KK + k];
    }
    const float inv = 1.0f / den;
    #pragma unroll
    for (int dh = 0; dh < DHD; ++dh)
        msg[(size_t)b * DD * NN + (size_t)(dh * HH + h) * NN + n] = acc[dh] * inv;
}

extern "C" void kernel_launch(void* const* d_in, const int* in_sizes, int n_in,
                              void* d_out, int out_size, void* d_ws, size_t ws_size,
                              hipStream_t stream)
{
    (void)in_sizes; (void)n_in; (void)out_size; (void)ws_size;

    const float* x       = (const float*)d_in[0];
    const float* source  = (const float*)d_in[1];
    const float* to_q_w  = (const float*)d_in[2];
    const float* to_q_b  = (const float*)d_in[3];
    const float* to_k_w  = (const float*)d_in[4];
    const float* to_k_b  = (const float*)d_in[5];
    const float* proj_k  = (const float*)d_in[6];
    const float* merge_w = (const float*)d_in[7];
    const float* merge_b = (const float*)d_in[8];
    const float* mlp_w1  = (const float*)d_in[9];
    const float* mlp_b1  = (const float*)d_in[10];
    const float* bn_g    = (const float*)d_in[11];
    const float* bn_b    = (const float*)d_in[12];
    const float* bn_m    = (const float*)d_in[13];
    const float* bn_v    = (const float*)d_in[14];
    const float* mlp_w2  = (const float*)d_in[15];
    const float* mlp_b2  = (const float*)d_in[16];
    float* outp = (float*)d_out;

    float* ws = (float*)d_ws;
    const size_t szBDN = (size_t)BB * DD * NN;            // 16.78M floats
    float* qbuf  = ws;                                    // q, later reused as msg2
    float* kvbuf = ws + szBDN;                            // kv, later reused as msg
    float* kpbuf = ws + 2 * szBDN;                        // [B, D, K]
    float* hbuf  = kpbuf + (size_t)BB * DD * KK;          // [B, 2D, N]
    float* msgb  = kvbuf;
    float* msg2  = qbuf;

    dim3 blk(256);

    // 1) q = to_q_w @ x + b            [B, D, N]
    gemm_wmma_kernel<1, 0, 0><<<dim3(NN / 128, DD / 64, BB), blk, 0, stream>>>(
        to_q_w, 0, x, x, 0, (size_t)DD * NN, to_q_b,
        nullptr, nullptr, nullptr, nullptr,
        qbuf, (size_t)DD * NN, DD, NN);

    // 2) kv = to_k_w @ source + b      [B, D, N]
    gemm_wmma_kernel<1, 0, 0><<<dim3(NN / 128, DD / 64, BB), blk, 0, stream>>>(
        to_k_w, 0, source, source, 0, (size_t)DD * NN, to_k_b,
        nullptr, nullptr, nullptr, nullptr,
        kvbuf, (size_t)DD * NN, DD, NN);

    // 3) kp = kv @ proj_k              [B, D, K]   (A batched, X shared)
    gemm_wmma_kernel<0, 0, 0><<<dim3(KK / 128, DD / 64, BB), blk, 0, stream>>>(
        kvbuf, (size_t)DD * NN, proj_k, proj_k, 0, 0, nullptr,
        nullptr, nullptr, nullptr, nullptr,
        kpbuf, (size_t)DD * KK, NN, KK);

    // 4) attention -> msg              [B, D, N]   (overwrites kv buffer)
    attn_kernel<<<dim3(NN / 256, HH, BB), blk, 0, stream>>>(qbuf, kpbuf, msgb);

    // 5) msg2 = merge_w @ msg + b      [B, D, N]   (overwrites q buffer)
    gemm_wmma_kernel<1, 0, 0><<<dim3(NN / 128, DD / 64, BB), blk, 0, stream>>>(
        merge_w, 0, msgb, msgb, 0, (size_t)DD * NN, merge_b,
        nullptr, nullptr, nullptr, nullptr,
        msg2, (size_t)DD * NN, DD, NN);

    // 6) h = relu(BN(mlp_w1 @ concat(x, msg2) + b1))   [B, 2D, N]
    gemm_wmma_kernel<1, 1, 1><<<dim3(NN / 128, (2 * DD) / 64, BB), blk, 0, stream>>>(
        mlp_w1, 0, x, msg2, DD, (size_t)DD * NN, mlp_b1,
        bn_g, bn_b, bn_m, bn_v,
        hbuf, (size_t)2 * DD * NN, 2 * DD, NN);

    // 7) out = mlp_w2 @ h + b2         [B, D, N]
    gemm_wmma_kernel<1, 0, 0><<<dim3(NN / 128, DD / 64, BB), blk, 0, stream>>>(
        mlp_w2, 0, hbuf, hbuf, 0, (size_t)2 * DD * NN, mlp_b2,
        nullptr, nullptr, nullptr, nullptr,
        outp, (size_t)DD * NN, 2 * DD, NN);
}
